// ScoreAggregation_4045859193723
// MI455X (gfx1250) — matrix operand order repaired and checked by reading, not compiled, not run on hardware
//
#include <hip/hip_runtime.h>
#include <hip/hip_bf16.h>

typedef __attribute__((ext_vector_type(16))) _Float16 v16h;
typedef __attribute__((ext_vector_type(8)))  float    v8f;

#define NN 8192
#define TT 4
#define EE 131072
#define ETOT (TT*EE)
#define DD 32
#define HH 4
#define WDP2 (DD+2)
#define NEG_SLOPE 0.2f
#define MAXDEG 512

// CDNA5 async global->LDS staging path (gfx1250). Guarded so the source
// cannot regress if this toolchain does not declare the builtins.
// The builtin's params are '__device__ int *' (addrspace(1)) and the LDS
// counterpart (addrspace(3)) per the round-2 diagnostic.
#if __has_builtin(__builtin_amdgcn_global_load_async_to_lds_b32) && \
    __has_builtin(__builtin_amdgcn_s_wait_asynccnt)
#define HAVE_ASYNC_LDS 1
typedef __attribute__((address_space(1))) int gas_int;
typedef __attribute__((address_space(3))) int las_int;
#endif

// ---------------------------------------------------------------------------
// K0: zero counts + cursors
// ---------------------------------------------------------------------------
__global__ void k_init(unsigned* counts, unsigned* cursor) {
    int i = blockIdx.x * blockDim.x + threadIdx.x;
    if (i < NN) { counts[i] = 0u; cursor[i] = 0u; }
}

// ---------------------------------------------------------------------------
// K1: head scalars.  emb_term = edge_type_emb(4x32) @ W_emb(32x4) via ONE
// v_wmma_f32_16x16x32_f16 (K=32 == D exactly), executed by wave 0.
// Also w_src/w_tgt extraction and deterministic block-sum of scores.
// ws layout (floats): [0:4) w_src, [4:8) w_tgt, [8:24) emb_term[h*4+t], [24] Stot
// ---------------------------------------------------------------------------
__global__ void k_scalars(const float* __restrict__ scores,
                          const float* __restrict__ emb,
                          const float* __restrict__ aw,
                          float* __restrict__ wsf) {
    __shared__ float red[256];
    const int tid = threadIdx.x;

    if (tid < 32) {                       // wave 0: EXEC all-ones -> WMMA legal
        const int lane = tid;
        const int m = lane & 15;
        const int ka = (lane < 16) ? 0 : 8;    // A 16-bit layout half-select
        const int kb = (lane < 16) ? 0 : 8;    // B 16-bit layout half-select
        v16h a, b;
        for (int j = 0; j < 8; ++j) {
            // A (16x32, rows = edge type t): VGPR j holds K pair
            int kA = (j < 4) ? (ka + 2*j) : (16 + ka + 2*(j-4));
            float a0 = (m < TT) ? emb[m*DD + kA]     : 0.0f;
            float a1 = (m < TT) ? emb[m*DD + kA + 1] : 0.0f;
            a[2*j]   = (_Float16)a0;
            a[2*j+1] = (_Float16)a1;
            // B (32x16, cols = head h): VGPR j holds same K pair as A
            int kB = (j < 4) ? (kb + 2*j) : (16 + kb + 2*(j-4));
            float b0 = (m < HH) ? aw[m*WDP2 + 1 + kB]     : 0.0f;
            float b1 = (m < HH) ? aw[m*WDP2 + 1 + kB + 1] : 0.0f;
            b[2*j]   = (_Float16)b0;
            b[2*j+1] = (_Float16)b1;
        }
        v8f c = {};
        c = __builtin_amdgcn_wmma_f32_16x16x32_f16(false, a, false, b,
                                                   (short)0, c, false, false);
        // D[t][h]: VGPR r holds M=r rows for lanes 0..15 (N = lane)
        if (lane < HH) {
            for (int t = 0; t < TT; ++t) wsf[8 + lane*TT + t] = c[t];
        }
    } else if (tid < 64) {
        int h = tid - 32;
        if (h < HH) {
            wsf[h]     = aw[h*WDP2 + 0];        // w_src
            wsf[4 + h] = aw[h*WDP2 + DD + 1];   // w_tgt
        }
    }

    // deterministic sum of scores (8192 / 256 = 32 per thread)
    float s = 0.0f;
    for (int i = tid; i < NN; i += 256) s += scores[i];
    red[tid] = s;
    __syncthreads();
    for (int off = 128; off > 0; off >>= 1) {
        if (tid < off) red[tid] += red[tid + off];
        __syncthreads();
    }
    if (tid == 0) wsf[24] = red[0];
}

// ---------------------------------------------------------------------------
// K2: per-row edge counting
// ---------------------------------------------------------------------------
__global__ void k_count(const int* __restrict__ ei, unsigned* __restrict__ counts) {
    int e = blockIdx.x * blockDim.x + threadIdx.x;
    if (e < ETOT) {
        int t = e >> 17;          // e / EE  (EE = 2^17)
        int r = e & (EE - 1);
        int src = ei[t*2*EE + r];
        atomicAdd(&counts[src], 1u);
    }
}

// ---------------------------------------------------------------------------
// K3: exclusive prefix scan of 8192 counts, single block x 1024 threads
// ---------------------------------------------------------------------------
__global__ void k_scan(const unsigned* __restrict__ counts,
                       unsigned* __restrict__ offs) {
    __shared__ unsigned part[1024];
    const int tid = threadIdx.x;
    unsigned vals[8], sum = 0;
    const int base = tid * 8;
    for (int i = 0; i < 8; ++i) { vals[i] = counts[base + i]; sum += vals[i]; }
    part[tid] = sum;
    __syncthreads();
    for (int off = 1; off < 1024; off <<= 1) {
        unsigned v = (tid >= off) ? part[tid - off] : 0u;
        __syncthreads();
        part[tid] += v;
        __syncthreads();
    }
    unsigned run = part[tid] - sum;           // exclusive base for this thread
    for (int i = 0; i < 8; ++i) { offs[base + i] = run; run += vals[i]; }
    if (tid == 1023) offs[NN] = part[1023];
}

// ---------------------------------------------------------------------------
// K4: CSR fill, packed key = (tgt<<2) | type
// ---------------------------------------------------------------------------
__global__ void k_fill(const int* __restrict__ ei,
                       const unsigned* __restrict__ offs,
                       unsigned* __restrict__ cursor,
                       unsigned* __restrict__ bucket) {
    int e = blockIdx.x * blockDim.x + threadIdx.x;
    if (e < ETOT) {
        int t = e >> 17;
        int r = e & (EE - 1);
        int src = ei[t*2*EE + r];
        int tgt = ei[t*2*EE + EE + r];
        unsigned pos = offs[src] + atomicAdd(&cursor[src], 1u);
        bucket[pos] = ((unsigned)tgt << 2) | (unsigned)t;
    }
}

// ---------------------------------------------------------------------------
// K5: one block (128 thr) per row: async-stage cells to LDS, sort, merge
// duplicates (integer, order-independent), closed-form sparse softmax,
// A@s, mean over heads.
// ---------------------------------------------------------------------------
__global__ __launch_bounds__(128)
void k_rows(const float* __restrict__ scores,
            const unsigned* __restrict__ bucket,
            const unsigned* __restrict__ offs,
            const float* __restrict__ wsf,
            float* __restrict__ out) {
    __shared__ unsigned key[MAXDEG];
    __shared__ unsigned cnt[MAXDEG];
    __shared__ float    sv [MAXDEG];
    __shared__ float    lv [MAXDEG];
    __shared__ float    red[128];

    const int tid = threadIdx.x;
    const int row = blockIdx.x;
    const unsigned o0 = offs[row];
    const unsigned o1 = offs[row + 1];
    int deg = (int)(o1 - o0);
    if (deg > MAXDEG) deg = MAXDEG;           // statistically unreachable

    __builtin_prefetch(bucket + o0, 0, 1);    // -> global_prefetch_b8

#ifdef HAVE_ASYNC_LDS
    // CDNA5 direct global->LDS async copy (ASYNCcnt-tracked), overlapped
    // with the sentinel fill; no VGPR round-trip.
    for (int c = tid; c < MAXDEG; c += 128) {
        if (c < deg) {
            __builtin_amdgcn_global_load_async_to_lds_b32(
                (gas_int*)(const void*)(bucket + o0 + c),
                (las_int*)(void*)&key[c], 0, 0);
        } else {
            key[c] = 0xFFFFFFFFu;
        }
    }
    __builtin_amdgcn_s_wait_asynccnt(0);
    __syncthreads();
#else
    for (int c = tid; c < MAXDEG; c += 128)
        key[c] = (c < deg) ? bucket[o0 + c] : 0xFFFFFFFFu;
    __syncthreads();
#endif

    // bitonic sort of 512 keys (sentinels sink to the end)
    for (int k = 2; k <= MAXDEG; k <<= 1) {
        for (int j = k >> 1; j > 0; j >>= 1) {
            for (int c = tid; c < MAXDEG; c += 128) {
                int ixj = c ^ j;
                if (ixj > c) {
                    unsigned av = key[c], bv = key[ixj];
                    bool up = ((c & k) == 0);
                    if ((av > bv) == up) { key[c] = bv; key[ixj] = av; }
                }
            }
            __syncthreads();
        }
    }

    // merge duplicate targets: run-start cells collect packed per-type counts
    for (int c = tid; c < MAXDEG; c += 128) cnt[c] = 0u;
    __syncthreads();
    for (int c = tid; c < deg; c += 128) {
        unsigned tg = key[c] >> 2;
        if (c == 0 || (key[c-1] >> 2) != tg) {
            unsigned pk = 0u;
            int r = c;
            while (r < deg && (key[r] >> 2) == tg) { pk += 1u << (8u * (key[r] & 3u)); ++r; }
            cnt[c] = pk;
            sv[c]  = scores[tg];
        }
    }
    __syncthreads();

    const float si   = scores[row];
    const float Stot = wsf[24];
    float acc = 0.0f;                          // only thread 0's copy used

    for (int h = 0; h < HH; ++h) {
        const float wsc = wsf[h];
        const float wtg = wsf[4 + h];
        const float e0 = wsf[8 + h*TT + 0];
        const float e1 = wsf[8 + h*TT + 1];
        const float e2 = wsf[8 + h*TT + 2];
        const float e3 = wsf[8 + h*TT + 3];

        // pass 1: values + leaky relu + max (zeros included via init 0)
        float lmax = 0.0f;
        for (int c = tid; c < deg; c += 128) {
            unsigned pk = cnt[c];
            if (pk) {
                float c0 = (float)( pk         & 255u);
                float c1 = (float)((pk >>  8u) & 255u);
                float c2 = (float)((pk >> 16u) & 255u);
                float c3 = (float)((pk >> 24u) & 255u);
                float kf = c0 + c1 + c2 + c3;
                float v  = kf * (wsc * si + wtg * sv[c])
                         + c0*e0 + c1*e1 + c2*e2 + c3*e3;
                float l  = (v >= 0.0f) ? v : NEG_SLOPE * v;
                lv[c] = l;
                lmax = fmaxf(lmax, l);
            }
        }
        red[tid] = lmax;
        __syncthreads();
        for (int s = 64; s > 0; s >>= 1) {
            if (tid < s) red[tid] = fmaxf(red[tid], red[tid + s]);
            __syncthreads();
        }
        const float m  = red[0];
        const float em = __expf(-m);
        __syncthreads();

        // pass 2: Z and numerator deltas
        float sa = 0.0f, sb = 0.0f;
        for (int c = tid; c < deg; c += 128) {
            if (cnt[c]) {
                float d = __expf(lv[c] - m) - em;
                sa += d;
                sb += d * sv[c];
            }
        }
        red[tid] = sa;
        __syncthreads();
        for (int s = 64; s > 0; s >>= 1) {
            if (tid < s) red[tid] += red[tid + s];
            __syncthreads();
        }
        const float A = red[0];
        __syncthreads();
        red[tid] = sb;
        __syncthreads();
        for (int s = 64; s > 0; s >>= 1) {
            if (tid < s) red[tid] += red[tid + s];
            __syncthreads();
        }
        if (tid == 0) {
            float Z   = (float)NN * em + A;
            float num = em * Stot + red[0];
            acc += num / Z;
        }
        __syncthreads();
    }

    if (tid == 0) out[row] = acc * (1.0f / (float)HH);
}

// ---------------------------------------------------------------------------
extern "C" void kernel_launch(void* const* d_in, const int* in_sizes, int n_in,
                              void* d_out, int out_size, void* d_ws, size_t ws_size,
                              hipStream_t stream) {
    const float* scores = (const float*)d_in[0];
    const float* emb    = (const float*)d_in[1];
    const int*   ei     = (const int*)  d_in[2];
    const float* aw     = (const float*)d_in[3];
    float*       out    = (float*)d_out;

    float*    wsf    = (float*)d_ws;                 // 32 floats of scalars
    unsigned* counts = (unsigned*)d_ws + 32;
    unsigned* cursor = counts + NN;
    unsigned* offs   = cursor + NN;                  // NN+1
    unsigned* bucket = offs + (NN + 1);              // ETOT

    k_init   <<<(NN + 255)/256, 256, 0, stream>>>(counts, cursor);
    k_scalars<<<1, 256, 0, stream>>>(scores, emb, aw, wsf);
    k_count  <<<(ETOT + 255)/256, 256, 0, stream>>>(ei, counts);
    k_scan   <<<1, 1024, 0, stream>>>(counts, offs);
    k_fill   <<<(ETOT + 255)/256, 256, 0, stream>>>(ei, offs, cursor, bucket);
    k_rows   <<<NN, 128, 0, stream>>>(scores, bucket, offs, wsf, out);
}
